// LocalMaskedMHA_72765335929373
// MI455X (gfx1250) — matrix operand order repaired and checked by reading, not compiled, not run on hardware
//
#include <hip/hip_runtime.h>

#define DEVFN __device__ __forceinline__

typedef __attribute__((ext_vector_type(16))) __bf16 v16bf;
typedef __attribute__((ext_vector_type(8)))  float  v8f;

constexpr int EMBD = 512;
constexpr int H    = 16;
constexpr int D    = 32;          // head dim == WMMA K for bf16 16x16x32
constexpr int T    = 4608;
constexpr int BS   = 8;
constexpr int S    = 9;           // half window (W = 19)
constexpr int NK   = EMBD / 32;   // K-steps in the projection GEMMs

union Frag16 {
  unsigned int   u[8];
  unsigned short h[16];
  v16bf          v;
};

DEVFN unsigned short f2bf(float f) {
  unsigned int u = __float_as_uint(f);
  u += 0x7FFFu + ((u >> 16) & 1u);          // round-to-nearest-even
  return (unsigned short)(u >> 16);
}

DEVFN v8f wmma_bf16(const Frag16& a, const Frag16& b, v8f c) {
  // (neg_a, A, neg_b, B, c_mod, C, reuse_a, reuse_b)
  return __builtin_amdgcn_wmma_f32_16x16x32_bf16(false, a.v, false, b.v,
                                                 (short)0, c, false, false);
}

// A-matrix (bf16 16x32) pair base K for pair p (pairs are contiguous K,K+1):
// lanes 0-15: K = 0..15; lanes 16-31: K = 16..31
DEVFN int a_k0(int lane, int p) {
  return ((lane >> 4) << 4) + (p << 1);
}
// B-matrix (bf16 32x16): lanes 0-15 hold K=0..15, lanes 16-31 hold K=16..31
DEVFN int b_k0(int lane, int p) { return ((lane >> 4) << 4) + (p << 1); }

// ------------------------------------------------------------------ TDM
#if __has_builtin(__builtin_amdgcn_tensor_load_to_lds) && \
    __has_builtin(__builtin_amdgcn_s_wait_tensorcnt)
#define ATHENA_TDM 1
typedef unsigned int v4u __attribute__((ext_vector_type(4)));
typedef int          v4i __attribute__((ext_vector_type(4)));
typedef int          v8i __attribute__((ext_vector_type(8)));

DEVFN unsigned int lds_off(const void* p) {        // LDS aperture: low 32 bits
  return (unsigned int)(unsigned long long)p;
}

// 2D tensor-tile DMA into LDS (row-major, no padding, no iterate/gather).
// D# layout per CDNA5 ISA ch.8: group0 = {count,lds,global,type}, group1 holds
// data_size, tensor_dim0/1, tile_dim0/1, tensor_dim0_stride.
DEVFN void tdm_load_2d(unsigned int lds_byte, const void* gaddr,
                       unsigned int dsize_code,           // 0:1B 1:2B 2:4B
                       unsigned int tile_w, unsigned int tile_h,
                       unsigned int tensor_w, unsigned int tensor_h,
                       unsigned long long row_stride_elems) {
  unsigned long long ga = (unsigned long long)gaddr;
  v4u g0;
  g0[0] = 1u;                                              // count=1, user D#
  g0[1] = lds_byte;                                        // lds_addr
  g0[2] = (unsigned int)ga;                                // global_addr lo
  g0[3] = (unsigned int)((ga >> 32) & 0x01FFFFFFu) | (2u << 30); // hi + type=2
  v8i g1;
  g1[0] = (int)(dsize_code << 16);                         // data_size
  g1[1] = (int)(tensor_w << 16);                           // tensor_dim0[15:0]
  g1[2] = (int)((tensor_w >> 16) | (tensor_h << 16));      // d0 hi | d1 lo
  g1[3] = (int)((tensor_h >> 16) | (tile_w << 16));        // d1 hi | tile_dim0
  g1[4] = (int)(tile_h & 0xFFFFu);                         // tile_dim1 (d2=0)
  g1[5] = (int)(unsigned int)(row_stride_elems);           // dim0_stride lo
  g1[6] = (int)(unsigned int)((row_stride_elems >> 32) & 0xFFFFull);
  g1[7] = 0;
  v4i z4 = {0, 0, 0, 0};
#if __clang_major__ >= 23
  v8i z8 = {0, 0, 0, 0, 0, 0, 0, 0};
  __builtin_amdgcn_tensor_load_to_lds(g0, g1, z4, z4, z8, 0);
#else
  __builtin_amdgcn_tensor_load_to_lds(g0, g1, z4, z4, 0);
#endif
}
#endif  // ATHENA_TDM

// ---------------------------------------------------------------- weights f32 -> bf16
__global__ void wconv_kernel(const float* __restrict__ w,
                             unsigned short* __restrict__ o, int n) {
  int i = blockIdx.x * 256 + threadIdx.x;
  if (i < n) o[i] = f2bf(__builtin_nontemporal_load(&w[i]));
}

// ---------------------------------------------------------------- tiled WMMA GEMM
// out[o,t] = sum_c W[o,c] * X[c,t]  (+bias, *scale)
// MODE 0: X = q (f32, [b][c][t]),  dst = head-layout bf16 [b*H+h][t][d]
// MODE 1: X = attn_out (bf16, head layout), dst = f32 [b][o][t] (NT stores)
// TDM path: double-buffered LDS, descriptors for step i+1 issued before
// waiting on step i (wait tensorcnt<=2: two in-order descriptors per step).
template <int MODE>
__launch_bounds__(256)
__global__ void gemm_kernel(const unsigned short* __restrict__ Wbf,
                            const float* __restrict__ bias,
                            const float* __restrict__ xq,
                            const unsigned short* __restrict__ xao,
                            unsigned short* __restrict__ dst_h,
                            float* __restrict__ dst_f,
                            float scale) {
  __shared__ unsigned short As[2][64][32];                 // [buf][m][k]
  __shared__ unsigned short Bs[2][128][32];                // [buf][n][k]
  __shared__ float Bf[(MODE == 0) ? 2 : 1][(MODE == 0) ? 32 : 1]
                     [(MODE == 0) ? 128 : 1];              // f32 staging

  const int tid  = threadIdx.x;
  const int lane = tid & 31;
  const int wave = tid >> 5;
  const int wm   = wave >> 1;                              // 4 M-tiles
  const int wn   = wave & 1;                               // 2 N-strips of 64
  const int b    = blockIdx.z;
  const int m0   = blockIdx.y * 64;
  const int t0   = blockIdx.x * 128;

  v8f acc[4];
#pragma unroll
  for (int j = 0; j < 4; ++j) acc[j] = {};

  auto compute_step = [&](int buf) {
    Frag16 a;
#pragma unroll
    for (int p = 0; p < 8; ++p)
      a.u[p] =
          *(const unsigned int*)&As[buf][(wm << 4) + (lane & 15)][a_k0(lane, p)];
#pragma unroll
    for (int j = 0; j < 4; ++j) {
      Frag16 bf;
#pragma unroll
      for (int p = 0; p < 8; ++p)
        bf.u[p] = *(const unsigned int*)
            &Bs[buf][(wn << 6) + (j << 4) + (lane & 15)][b_k0(lane, p)];
      acc[j] = wmma_bf16(a, bf, acc[j]);
    }
  };

#ifdef ATHENA_TDM
  auto issue_step = [&](int i) {
    const int buf = i & 1;
    const int kk  = i * 32;
    // A panel: 64x32 bf16 tile of W, row stride EMBD
    tdm_load_2d(lds_off(&As[buf][0][0]), &Wbf[(size_t)m0 * EMBD + kk],
                1, 32, 64, 32, 64, EMBD);
    if constexpr (MODE == 0) {
      // B panel: 32x128 f32 tile of q (rows = channels, stride T)
      tdm_load_2d(lds_off(&Bf[buf][0][0]), &xq[((size_t)b * EMBD + kk) * T + t0],
                  2, 128, 32, 128, 32, T);
    } else {
      // B panel: TDM does the [t][d] -> [n][k] re-layout directly
      tdm_load_2d(lds_off(&Bs[buf][0][0]),
                  &xao[((size_t)(b * H + (kk >> 5)) * T + t0) * D],
                  1, 32, 128, 32, 128, D);
    }
  };

  if (wave == 0) issue_step(0);
  for (int i = 0; i < NK; ++i) {
    const int buf = i & 1;
    if (wave == 0) {
      if (i + 1 < NK) {
        issue_step(i + 1);                       // run TDM ahead of WMMA
        __builtin_amdgcn_s_wait_tensorcnt(2);    // step i's pair complete
      } else {
        __builtin_amdgcn_s_wait_tensorcnt(0);    // drain
      }
    }
    __syncthreads();                             // buf data visible block-wide
    if constexpr (MODE == 0) {                   // LDS f32 -> LDS bf16 (+transpose)
      int k = tid >> 3, colb = (tid & 7) << 4;
#pragma unroll
      for (int i2 = 0; i2 < 16; ++i2)
        Bs[buf][colb + i2][k] = f2bf(Bf[buf][k][colb + i2]);
      __syncthreads();
    }
    compute_step(buf);
    __syncthreads();                             // done reading buf
  }
#else
  for (int kk = 0; kk < EMBD; kk += 32) {
    { // stage A panel (64x32 bf16), 16B per thread
      int row = tid >> 2, col = (tid & 3) << 3;
      *(uint4*)&As[0][row][col] =
          *(const uint4*)&Wbf[(size_t)(m0 + row) * EMBD + kk + col];
    }
    { // stage B panel (32x128 -> stored [n][k])
      int k = tid >> 3, colb = (tid & 7) << 4;
      if (MODE == 0) {
        const float* src = &xq[((size_t)b * EMBD + kk + k) * T + t0 + colb];
        if (kk + 32 < EMBD)
          __builtin_prefetch(src + (size_t)32 * T, 0, 1);  // global_prefetch_b8
#pragma unroll
        for (int i = 0; i < 16; ++i) Bs[0][colb + i][k] = f2bf(src[i]);
      } else {
        int c = kk + k;
        const unsigned short* base =
            &xao[((size_t)(b * H + (c >> 5)) * T + (t0 + colb)) * D + (c & 31)];
#pragma unroll
        for (int i = 0; i < 16; ++i) Bs[0][colb + i][k] = base[(size_t)i * D];
      }
    }
    __syncthreads();
    compute_step(0);
    __syncthreads();
  }
#endif

#pragma unroll
  for (int j = 0; j < 4; ++j) {
#pragma unroll
    for (int r = 0; r < 8; ++r) {
      int o = m0 + (wm << 4) + ((lane >> 4) << 3) + r;
      int t = t0 + (wn << 6) + (j << 4) + (lane & 15);
      float v = acc[j][r] + bias[o];
      if (MODE == 0) {
        dst_h[((size_t)(b * H + (o >> 5)) * T + t) * D + (o & 31)] =
            f2bf(v * scale);
      } else {
        // write-once output: stream past L2 so bf16 intermediates stay resident
        __builtin_nontemporal_store(
            v, &dst_f[(size_t)b * EMBD * T + (size_t)o * T + t]);
      }
    }
  }
}

// ---------------------------------------------------------------- banded attention
// One wave per 16-query tile; keys from 3 aligned 16-key tiles (48, padded to 64).
// All LDS regions are per-wave; LDS ops are in-order within a wave. The V-tile
// TDM is waited only right before PV, overlapping the DMA with QK^T + softmax.
__launch_bounds__(128)
__global__ void attn_kernel(const unsigned short* __restrict__ qh,
                            const unsigned short* __restrict__ kh,
                            const unsigned short* __restrict__ vh,
                            unsigned short* __restrict__ ao) {
  __shared__ unsigned short P[4][16][64];    // probabilities, bf16
  __shared__ unsigned short Vs[4][64][32];   // staged V tile per wave
  const int tid  = threadIdx.x;
  const int lane = tid & 31;
  const int wave = tid >> 5;
  const int qt   = blockIdx.x * 4 + wave;    // query tile
  const int n    = blockIdx.y;               // b*H + h
  const size_t nT = (size_t)n * T;
  const int start = (qt - 1) << 4;           // first key row of the 64-row tile

  // ---- kick off V tile DMA (64 keys x 32 d)
#ifdef ATHENA_TDM
  {
    const int rb = (start < 0) ? -start : 0;          // leading invalid rows
    for (int i = lane; i < rb * 16; i += 32)          // zero rows (qt==0 only)
      ((unsigned int*)&Vs[wave][0][0])[i] = 0u;
    const int srow = start + rb;
    // upper edge handled by tensor_dim1 clamp: OOB rows read back as zero
    tdm_load_2d(lds_off(&Vs[wave][rb][0]), &vh[(nT + srow) * D],
                1, 32, 64 - rb, 32, (unsigned)(T - srow), D);
  }
#else
  for (int rr = lane; rr < 64; rr += 32) {
    int key = start + rr;
    uint4* dstp = (uint4*)&Vs[wave][rr][0];
    if (key >= 0 && key < T) {
      const uint4* src = (const uint4*)&vh[(nT + key) * D];
#pragma unroll
      for (int i = 0; i < 4; ++i) dstp[i] = src[i];
    } else {
      uint4 z = {0u, 0u, 0u, 0u};
#pragma unroll
      for (int i = 0; i < 4; ++i) dstp[i] = z;
    }
  }
#endif

  // ---- Q fragment: A layout straight from global (d contiguous)
  Frag16 aq;
  {
    int t = (qt << 4) + (lane & 15);
    const unsigned short* row = &qh[(nT + t) * D];
#pragma unroll
    for (int p = 0; p < 8; ++p)
      aq.u[p] = *(const unsigned int*)&row[a_k0(lane, p)];
  }

  // ---- scores: 3 WMMAs, K = d = 32
  v8f sc[3];
#pragma unroll
  for (int i = 0; i < 3; ++i) {
    int key = start + (i << 4) + (lane & 15);
    Frag16 bk;
    if (key >= 0 && key < T) {
      const unsigned short* row = &kh[(nT + key) * D];
#pragma unroll
      for (int p = 0; p < 8; ++p)
        bk.u[p] = *(const unsigned int*)&row[b_k0(lane, p)];
    } else {
#pragma unroll
      for (int p = 0; p < 8; ++p) bk.u[p] = 0u;
    }
    v8f z = {};
    sc[i] = wmma_bf16(aq, bk, z);
  }

  // ---- window mask + softmax. C layout: lane holds key-col (lane&15) of rows
  // m = (lane>>4)*8 + r; row reductions are xor-shuffles within 16-lane halves.
  float pv[3][8];
#pragma unroll
  for (int r = 0; r < 8; ++r) {
    int tg = (qt << 4) + ((lane >> 4) << 3) + r;
    float mx = -3.0e38f;
#pragma unroll
    for (int i = 0; i < 3; ++i) {
      int kg = start + (i << 4) + (lane & 15);
      bool ok = (kg >= 0) && (kg < T) && (kg >= tg - S) && (kg <= tg + S);
      float x = ok ? sc[i][r] : -3.0e38f;
      pv[i][r] = x;
      mx = fmaxf(mx, x);
    }
#pragma unroll
    for (int off = 1; off < 16; off <<= 1)
      mx = fmaxf(mx, __shfl_xor(mx, off, 32));
    float sum = 0.f;
#pragma unroll
    for (int i = 0; i < 3; ++i) {
      float e = __expf(pv[i][r] - mx);
      pv[i][r] = e;
      sum += e;
    }
#pragma unroll
    for (int off = 1; off < 16; off <<= 1) sum += __shfl_xor(sum, off, 32);
    float inv = 1.0f / sum;
    int m = ((lane >> 4) << 3) + r;
#pragma unroll
    for (int i = 0; i < 3; ++i)
      P[wave][m][(i << 4) + (lane & 15)] = f2bf(pv[i][r] * inv);
    P[wave][m][48 + (lane & 15)] = 0;        // zero pad cols 48..63
  }

#ifdef ATHENA_TDM
  __builtin_amdgcn_s_wait_tensorcnt(0);      // V tile ready (overlapped above)
#endif

  // ---- PV: P(16x64) x V(64x32) = 4 WMMAs (V fragments from LDS)
  v8f oacc[2];
  oacc[0] = {}; oacc[1] = {};
#pragma unroll
  for (int s = 0; s < 2; ++s) {
    Frag16 ap;
    const unsigned short* prow = &P[wave][lane & 15][0];
#pragma unroll
    for (int p = 0; p < 8; ++p)
      ap.u[p] = *(const unsigned int*)&prow[(s << 5) + a_k0(lane, p)];
#pragma unroll
    for (int dt = 0; dt < 2; ++dt) {
      Frag16 bv;
#pragma unroll
      for (int hi = 0; hi < 16; ++hi) {
        int k = (s << 5) + ((lane >> 4) << 4) + hi;
        bv.h[hi] = Vs[wave][k][(dt << 4) + (lane & 15)];
      }
      oacc[dt] = wmma_bf16(ap, bv, oacc[dt]);
    }
  }

#pragma unroll
  for (int dt = 0; dt < 2; ++dt)
#pragma unroll
    for (int r = 0; r < 8; ++r) {
      int t = (qt << 4) + ((lane >> 4) << 3) + r;
      ao[(nT + t) * D + (dt << 4) + (lane & 15)] = f2bf(oacc[dt][r]);
    }
}

// ---------------------------------------------------------------- mask tail
__global__ void mask_tail_kernel(const unsigned char* __restrict__ qm,
                                 float* __restrict__ out) {
  int i = blockIdx.x * 256 + threadIdx.x;
  if (i < BS * T)
    __builtin_nontemporal_store(qm[i] ? 1.0f : 0.0f, &out[i]);
}

extern "C" void kernel_launch(void* const* d_in, const int* in_sizes, int n_in,
                              void* d_out, int out_size, void* d_ws, size_t ws_size,
                              hipStream_t stream) {
  (void)in_sizes; (void)n_in; (void)out_size; (void)ws_size;
  const float*         q     = (const float*)d_in[0];
  const unsigned char* qmask = (const unsigned char*)d_in[1];
  const float* Wq = (const float*)d_in[2];
  const float* bq = (const float*)d_in[3];
  const float* Wk = (const float*)d_in[4];
  const float* bk = (const float*)d_in[5];
  const float* Wv = (const float*)d_in[6];
  const float* bv = (const float*)d_in[7];
  const float* Wp = (const float*)d_in[8];
  const float* bp = (const float*)d_in[9];

  // workspace carve-up (bf16 intermediates: total ~155 MB -> L2 resident)
  char*  ws  = (char*)d_ws;
  size_t off = 0;
  auto carve = [&](size_t bytes) {
    void* p = ws + off;
    off = (off + bytes + 255) & ~(size_t)255;
    return p;
  };
  const size_t wsz = (size_t)EMBD * EMBD * sizeof(unsigned short);
  const size_t hsz = (size_t)BS * H * T * D * sizeof(unsigned short);
  unsigned short* wqb = (unsigned short*)carve(wsz);
  unsigned short* wkb = (unsigned short*)carve(wsz);
  unsigned short* wvb = (unsigned short*)carve(wsz);
  unsigned short* wpb = (unsigned short*)carve(wsz);
  unsigned short* qhb = (unsigned short*)carve(hsz);
  unsigned short* khb = (unsigned short*)carve(hsz);
  unsigned short* vhb = (unsigned short*)carve(hsz);
  unsigned short* aob = (unsigned short*)carve(hsz);

  const int wn = EMBD * EMBD;
  wconv_kernel<<<(wn + 255) / 256, 256, 0, stream>>>(Wq, wqb, wn);
  wconv_kernel<<<(wn + 255) / 256, 256, 0, stream>>>(Wk, wkb, wn);
  wconv_kernel<<<(wn + 255) / 256, 256, 0, stream>>>(Wv, wvb, wn);
  wconv_kernel<<<(wn + 255) / 256, 256, 0, stream>>>(Wp, wpb, wn);

  dim3 gg(T / 128, EMBD / 64, BS);
  const float scale = 0.17677669529663687f;   // 1/sqrt(32)
  gemm_kernel<0><<<gg, 256, 0, stream>>>(wqb, bq, q, nullptr, qhb, nullptr, scale);
  gemm_kernel<0><<<gg, 256, 0, stream>>>(wkb, bk, q, nullptr, khb, nullptr, 1.0f);
  gemm_kernel<0><<<gg, 256, 0, stream>>>(wvb, bv, q, nullptr, vhb, nullptr, 1.0f);

  attn_kernel<<<dim3(T / 64, BS * H), 128, 0, stream>>>(qhb, khb, vhb, aob);

  gemm_kernel<1><<<gg, 256, 0, stream>>>(wpb, bp, nullptr, aob, nullptr,
                                         (float*)d_out, 1.0f);

  mask_tail_kernel<<<(BS * T + 255) / 256, 256, 0, stream>>>(
      qmask, (float*)d_out + (size_t)BS * EMBD * T);
}
